// EMDLoss_5514738008419
// MI455X (gfx1250) — matrix operand order repaired and checked by reading, not compiled, not run on hardware
//
#include <hip/hip_runtime.h>
#include <hip/hip_bf16.h>

// ---------------------------------------------------------------------------
// EMD / Sinkhorn loss for B=16, C=64, H=W=48 (HW=2304) on gfx1250 (MI455X).
//
// Key points:
//  * M = 1 - Xn^T Yn  (16 x 2304 x 2304, fp32) must be written to d_out:
//    340 MB store -> the GEMM kernel is HBM-store bound at 23.3 TB/s.
//    v_wmma_f32_16x16x32_f16 supplies the ~750 TF/s needed to stay there.
//  * Sinkhorn matvecs use (M x)_i = sum(x) - Xn_i . (Yn x): O(C*HW) each,
//    never touching the 340 MB matrix. The reference loop is idempotent
//    (every iteration restarts from the constant c_), so one pass suffices.
//  * similarity = sum_ij u_i v_j exp(S-1) S is fused into the GEMM kernel
//    with deterministic per-wave partials (no float atomics).
// ---------------------------------------------------------------------------

#define BATCH 16
#define CHAN  64
#define HWN   2304          // 48*48
#define ITILES 144          // HWN/16
#define JBLKS  9            // HWN/256
#define STRIPS_PER_B (ITILES*JBLKS)   // 1296
#define NSTRIPS (BATCH*STRIPS_PER_B)  // 20736

typedef _Float16 v16h __attribute__((ext_vector_type(16)));
typedef _Float16 h8   __attribute__((ext_vector_type(8)));
typedef _Float16 h2   __attribute__((ext_vector_type(2)));
typedef float    v8f  __attribute__((ext_vector_type(8)));

// ---------------------------------------------------------------------------
// Kernel 1: L2-normalize along channel dim, convert to f16, store [B][HW][C]
// (row = spatial index, 64 contiguous f16 channels -> WMMA-fragment friendly)
// ---------------------------------------------------------------------------
__global__ __launch_bounds__(256) void k_normalize(const float* __restrict__ X,
                                                   const float* __restrict__ Y,
                                                   _Float16* __restrict__ Xn,
                                                   _Float16* __restrict__ Yn) {
  int idx = blockIdx.x * 256 + threadIdx.x;          // 2*B*HW threads
  const int total = BATCH * HWN;
  bool isY = idx >= total;
  int e = isY ? (idx - total) : idx;
  int b = e / HWN, i = e % HWN;
  const float* src = (isY ? Y : X) + (size_t)b * CHAN * HWN + i;
  _Float16* dst = (isY ? Yn : Xn) + ((size_t)b * HWN + i) * CHAN;

  float val[CHAN];
  float ss = 0.f;
#pragma unroll
  for (int c = 0; c < CHAN; ++c) {
    float x = src[(size_t)c * HWN];
    val[c] = x;
    ss += x * x;
  }
  float inv = 1.0f / fmaxf(sqrtf(ss), 1e-12f);
#pragma unroll
  for (int c = 0; c < CHAN; c += 2) {
    h2 p;
    p[0] = (_Float16)(val[c] * inv);
    p[1] = (_Float16)(val[c + 1] * inv);
    *(h2*)(dst + c) = p;
  }
}

// ---------------------------------------------------------------------------
// Kernel 2: Sinkhorn (factorized). One 1024-thread workgroup per batch.
//   (M x)_i   = sum(x) - Xn_i . (Yn x)
//   (M^T x)_j = sum(x) - Yn_j . (Xn x)
// ---------------------------------------------------------------------------
__device__ void matvec_chan(const _Float16* __restrict__ Mat,  // [HW][C]
                            const float* __restrict__ vec,     // nullptr => 1/HW
                            float* w, float* red, int tid) {
  const float inv_hw = 1.0f / HWN;
  int k = tid >> 4, g = tid & 15;                 // 64 channels x 16 groups
  float acc = 0.f;
  int j0 = g * (HWN / 16);
  for (int j = j0; j < j0 + (HWN / 16); ++j) {
    float vv = vec ? vec[j] : inv_hw;
    acc += (float)Mat[(size_t)j * CHAN + k] * vv;
  }
  red[tid] = acc;
  __syncthreads();
  if (g < 8) red[tid] += red[tid + 8];
  __syncthreads();
  if (g < 4) red[tid] += red[tid + 4];
  __syncthreads();
  if (g < 2) red[tid] += red[tid + 2];
  __syncthreads();
  if (g == 0) w[k] = red[tid] + red[tid + 1];
  __syncthreads();
}

__device__ float sum_vec(const float* __restrict__ vec, float* red, int tid) {
  float a = 0.f;
  for (int j = tid; j < HWN; j += 1024) a += vec[j];
  red[tid] = a;
  __syncthreads();
  for (int s = 512; s > 0; s >>= 1) {
    if (tid < s) red[tid] += red[tid + s];
    __syncthreads();
  }
  float r = red[0];
  __syncthreads();
  return r;
}

__device__ void apply_vec(const _Float16* __restrict__ Mat,  // [HW][C]
                          const float* w, float s, float num,
                          float* __restrict__ out, int tid) {
  const float eps = 1e-8f;
  for (int i = tid; i < HWN; i += 1024) {
    const _Float16* row = Mat + (size_t)i * CHAN;
    float acc = 0.f;
#pragma unroll
    for (int k = 0; k < CHAN; ++k) acc += (float)row[k] * w[k];
    out[i] = num / ((s - acc) + eps);
  }
}

__global__ __launch_bounds__(1024) void k_sinkhorn(const _Float16* __restrict__ Xn,
                                                   const _Float16* __restrict__ Yn,
                                                   float* __restrict__ u,
                                                   float* __restrict__ v) {
  __shared__ float w[CHAN];
  __shared__ float red[1024];
  int b = blockIdx.x, tid = threadIdx.x;
  const _Float16* Xb = Xn + (size_t)b * HWN * CHAN;
  const _Float16* Yb = Yn + (size_t)b * HWN * CHAN;
  float* ub = u + b * HWN;
  float* vb = v + b * HWN;
  const float inv_hw = 1.0f / HWN;

  // u = 1 / (M c + eps); sum(c) == 1
  matvec_chan(Yb, nullptr, w, red, tid);
  apply_vec(Xb, w, 1.0f, 1.0f, ub, tid);
  __syncthreads();
  // v = 1 / (M^T u + eps)
  float su = sum_vec(ub, red, tid);
  matvec_chan(Xb, ub, w, red, tid);
  apply_vec(Yb, w, su, 1.0f, vb, tid);
  __syncthreads();
  // u = r / (M v + eps)
  float sv = sum_vec(vb, red, tid);
  matvec_chan(Yb, vb, w, red, tid);
  apply_vec(Xb, w, sv, inv_hw, ub, tid);
  __syncthreads();
  // v = c / (M^T u + eps)
  float su2 = sum_vec(ub, red, tid);
  matvec_chan(Xb, ub, w, red, tid);
  apply_vec(Yb, w, su2, inv_hw, vb, tid);
}

// ---------------------------------------------------------------------------
// Kernel 3: WMMA GEMM. Each wave computes a 16x256 strip of S = Xn^T Yn,
// writes M = 1 - S, and accumulates sum(u_i v_j exp(S-1) S) for the strip.
// A-fragment per CDNA5 ISA 16-bit A layout (lane: M=l&15, K-half=l>>4;
// elems 0..7 -> K = kh*8+0..7, elems 8..15 -> K = 16+kh*8+0..7).
// B (32x16, K x N) loaded as lane: N=l&15, K-half=l>>4, same K order as A.
// ---------------------------------------------------------------------------
__global__ __launch_bounds__(256) void k_gemm_m(const _Float16* __restrict__ Xn,
                                                const _Float16* __restrict__ Yn,
                                                const float* __restrict__ u,
                                                const float* __restrict__ v,
                                                float* __restrict__ Mout,
                                                float* __restrict__ partials) {
  int wave = (blockIdx.x << 3) + (threadIdx.x >> 5);
  int lane = threadIdx.x & 31;
  int b = wave / STRIPS_PER_B;
  int rem = wave % STRIPS_PER_B;
  int it = rem / JBLKS, jb = rem % JBLKS;
  int i0 = it * 16, j0 = jb * 256;

  const _Float16* Xb = Xn + (size_t)b * HWN * CHAN;
  const _Float16* Yb = Yn + (size_t)b * HWN * CHAN;

  int m = lane & 15, kh = lane >> 4;

  // A fragments for K=0..31 and K=32..63 (reused across 16 j-tiles)
  const _Float16* arow = Xb + (size_t)(i0 + m) * CHAN;
  v16h a0, a1;
  {
    h8 lo = *(const h8*)(arow + kh * 8);
    h8 hi = *(const h8*)(arow + 16 + kh * 8);
#pragma unroll
    for (int t = 0; t < 8; ++t) { a0[t] = lo[t]; a0[8 + t] = hi[t]; }
  }
  {
    h8 lo = *(const h8*)(arow + 32 + kh * 8);
    h8 hi = *(const h8*)(arow + 48 + kh * 8);
#pragma unroll
    for (int t = 0; t < 8; ++t) { a1[t] = lo[t]; a1[8 + t] = hi[t]; }
  }

  const float* ub = u + b * HWN;
  const float* vb = v + b * HWN;
  int rbase = i0 + ((lane < 16) ? 0 : 8);
  float uu[8];
#pragma unroll
  for (int r = 0; r < 8; ++r) uu[r] = ub[rbase + r];

  float sim = 0.f;
  for (int jt = 0; jt < 16; ++jt) {
    int j = j0 + jt * 16 + m;                       // this lane's column
    const _Float16* brow = Yb + (size_t)j * CHAN + kh * 8;
    if (jt < 15)
      __builtin_prefetch(brow + 16 * CHAN, 0, 1);   // next j-tile (global_prefetch_b8)
    v16h b0, b1;
    {
      h8 lo = *(const h8*)(brow);
      h8 hi = *(const h8*)(brow + 16);
#pragma unroll
      for (int t = 0; t < 8; ++t) { b0[t] = lo[t]; b0[8 + t] = hi[t]; }
    }
    {
      h8 lo = *(const h8*)(brow + 32);
      h8 hi = *(const h8*)(brow + 48);
#pragma unroll
      for (int t = 0; t < 8; ++t) { b1[t] = lo[t]; b1[8 + t] = hi[t]; }
    }

    v8f acc = {};
    acc = __builtin_amdgcn_wmma_f32_16x16x32_f16(false, a0, false, b0,
                                                 (short)0, acc, false, false);
    acc = __builtin_amdgcn_wmma_f32_16x16x32_f16(false, a1, false, b1,
                                                 (short)0, acc, false, false);

    float vc = vb[j];
    size_t base = ((size_t)b * HWN + rbase) * HWN + j;
#pragma unroll
    for (int r = 0; r < 8; ++r) {
      float S = acc[r];
      Mout[base + (size_t)r * HWN] = 1.0f - S;      // M = 1 - S
      sim += uu[r] * vc * __expf(S - 1.0f) * S;     // u*exp(-M)*v*(1-M)
    }
  }

  // wave32 reduction, lane 0 writes deterministic per-strip partial
#pragma unroll
  for (int off = 16; off > 0; off >>= 1) sim += __shfl_xor(sim, off, 32);
  if (lane == 0) partials[wave] = sim;
}

// ---------------------------------------------------------------------------
// Kernel 4: per-batch reduction of strip partials -> similarity[b]
// ---------------------------------------------------------------------------
__global__ __launch_bounds__(256) void k_reduce_sim(const float* __restrict__ partials,
                                                    float* __restrict__ simout) {
  __shared__ float red[256];
  int b = blockIdx.x;
  float a = 0.f;
  for (int i = threadIdx.x; i < STRIPS_PER_B; i += 256)
    a += partials[b * STRIPS_PER_B + i];
  red[threadIdx.x] = a;
  __syncthreads();
  for (int s = 128; s > 0; s >>= 1) {
    if (threadIdx.x < s) red[threadIdx.x] += red[threadIdx.x + s];
    __syncthreads();
  }
  if (threadIdx.x == 0) simout[b] = red[0];
}

// ---------------------------------------------------------------------------
// Kernel 5: L_EMD = mean(2 - 2*similarity)
// ---------------------------------------------------------------------------
__global__ void k_final(const float* __restrict__ sim, float* __restrict__ out) {
  float a = (threadIdx.x < BATCH) ? (2.0f - 2.0f * sim[threadIdx.x]) : 0.f;
#pragma unroll
  for (int off = 16; off > 0; off >>= 1) a += __shfl_xor(a, off, 32);
  if (threadIdx.x == 0) out[0] = a * (1.0f / BATCH);
}

// ---------------------------------------------------------------------------
extern "C" void kernel_launch(void* const* d_in, const int* in_sizes, int n_in,
                              void* d_out, int out_size, void* d_ws, size_t ws_size,
                              hipStream_t stream) {
  const float* X = (const float*)d_in[0];
  const float* Y = (const float*)d_in[1];
  float* out = (float*)d_out;
  const size_t MSIZE = (size_t)BATCH * HWN * HWN;   // 84,934,656 floats

  // workspace layout (~9.82 MB total)
  char* ws = (char*)d_ws;
  const size_t HBYTES = (size_t)BATCH * HWN * CHAN * sizeof(_Float16); // 4,718,592
  _Float16* Xn = (_Float16*)ws;
  _Float16* Yn = (_Float16*)(ws + HBYTES);
  float* u        = (float*)(ws + 2 * HBYTES);
  float* v        = u + BATCH * HWN;
  float* partials = v + BATCH * HWN;                 // NSTRIPS floats

  k_normalize<<<(2 * BATCH * HWN) / 256, 256, 0, stream>>>(X, Y, Xn, Yn);
  k_sinkhorn<<<BATCH, 1024, 0, stream>>>(Xn, Yn, u, v);
  k_gemm_m<<<NSTRIPS / 8, 256, 0, stream>>>(Xn, Yn, u, v, out, partials);
  k_reduce_sim<<<BATCH, 256, 0, stream>>>(partials, out + MSIZE);
  k_final<<<1, 32, 0, stream>>>(out + MSIZE, out + MSIZE + BATCH);
}
